// StageNet_63806034149847
// MI455X (gfx1250) — compile-verified, matrix operand order checked by reading, not run on hardware
//
#include <hip/hip_runtime.h>
#include <hip/hip_bf16.h>
#include <cstdint>

// ---------------- problem constants ----------------
#define BB   256
#define TT   400
#define FF   76
#define HH   768
#define LL   8
#define KK   10
#define SS   128
#define GG   3088            // 4*H + 2*L
#define KA   864             // 77 + 769 = 846, padded to multiple of 32
#define CCH  96              // H / L
#define HK   (HH*KK)         // 7680, conv reduction dim
#define NTZ  196             // z-GEMM N tiles padded to multiple of 4 (193 -> 196)

typedef __attribute__((ext_vector_type(16))) __bf16 v16bf;
typedef __attribute__((ext_vector_type(8)))  float  v8f;

union Frag32 { uint4 u4[2]; v16bf v; };

__device__ __forceinline__ uint16_t f2bf(float x) {
    uint32_t u = __float_as_uint(x);
    uint32_t r = (u + 0x7FFFu + ((u >> 16) & 1u)) >> 16;   // RNE
    return (uint16_t)r;
}
__device__ __forceinline__ float sigm(float x) { return 1.0f / (1.0f + expf(-x)); }

// ---------------- zero init ----------------
__global__ void sn_zero_f32(float* __restrict__ p, int n) {
    int i = blockIdx.x * blockDim.x + threadIdx.x;
    if (i < n) p[i] = 0.0f;
}

// ---------------- weight swizzle into WMMA B-fragment order ----------------
// dst linear index i -> (ntile, kslab, lane, e); element = W[k][n] where
//   n = ntile*16 + (lane&15), half = lane>>4, pair = e>>1,
//   k = kslab*32 + (pair>>2)*16 + half*8 + (pair&3)*2 + (e&1)
// n >= Nreal produces zero padding (for padded N tiles).
__global__ void sn_swizzleB(const float* __restrict__ srcA,
                            const float* __restrict__ srcB,
                            uint16_t* __restrict__ dst,
                            int KS, int total, int Nreal, int mode) {
    int i = blockIdx.x * blockDim.x + threadIdx.x;
    if (i >= total) return;
    int e    = i & 15;
    int lane = (i >> 4) & 31;
    int rest = i >> 9;
    int ks   = rest % KS;
    int nt   = rest / KS;
    int n    = nt * 16 + (lane & 15);
    int half = lane >> 4;
    int pair = e >> 1;
    int k    = ks * 32 + (pair >> 2) * 16 + half * 8 + (pair & 3) * 2 + (e & 1);
    float val = 0.0f;
    if (n < Nreal) {
        if (mode == 0) {            // combined [kernel_w(77) ; rec_w(769) ; pad] x 3088
            if (k < 77)       val = srcA[(size_t)k * GG + n];
            else if (k < 846) val = srcB[(size_t)(k - 77) * GG + n];
        } else if (mode == 1) {     // conv_w (o=768, c=768, k=10): reduction idx = c*10+kk, n = o
            int c = k / KK, kk = k % KK;
            val = srcA[(size_t)n * HK + (size_t)c * KK + kk];
        } else if (mode == 2) {     // scale_w (768 x 128)
            val = srcA[(size_t)k * SS + n];
        } else {                    // rescale_w (128 x 768)
            val = srcA[(size_t)k * HH + n];
        }
    }
    dst[i] = f2bf(val);
}

__global__ void sn_biasz(const float* __restrict__ kb, const float* __restrict__ rb,
                         float* __restrict__ bz) {
    int i = blockIdx.x * blockDim.x + threadIdx.x;
    if (i < GG) bz[i] = kb[i] + rb[i];
}

// ---------------- per-step: load x_t into A matrix (bf16 row-major, lda=KA) ----------------
__global__ void sn_loadx(const float* __restrict__ x, uint16_t* __restrict__ Az, int t) {
    int b = blockIdx.x;
    int i = threadIdx.x;
    if (i < FF)       Az[(size_t)b * KA + i]   = f2bf(x[((size_t)b * TT + t) * FF + i]);
    else if (i == FF) Az[(size_t)b * KA + FF]  = f2bf(1.0f);   // ones col for kernel_w
    else if (i == FF + 1) Az[(size_t)b * KA + 845] = f2bf(1.0f); // ones col for rec_w
}

// ---------------- WMMA GEMM: one wave computes a 16(M) x 64(N) strip ----------------
// 4 N-tiles per wave: the A fragment is loaded once per K-slab and reused by
// 4 back-to-back v_wmma ops (2 A-loads + 8 B-loads per 4 WMMAs). Simple loop:
// the compiler unrolls and software-pipelines it without spilling.
// A: row-major bf16 (lda elems).  Bsw: fragment-swizzled bf16 (padded N tiles).
// MODE 0: Cf = acc+bias              (z GEMM; stores guarded by col < N)
// MODE 1: Cbf = bf16(relu(acc+bias)) (theme GEMM1)
// MODE 2: Cf = sigmoid(acc+bias)     (theme GEMM2)
// MODE 3: Cf = theme*(acc+bias)+hAdd (conv GEMM epilogue -> rnn)
template <int MODE>
__global__ void sn_gemm_bf16(const uint16_t* __restrict__ A, int lda,
                             const uint16_t* __restrict__ Bsw,
                             const float* __restrict__ bias,
                             float* __restrict__ Cf, uint16_t* __restrict__ Cbf,
                             int N, int KS,
                             const float* __restrict__ theme,
                             const float* __restrict__ hAdd) {
    const int lane  = threadIdx.x;          // wave32
    const int nt0   = blockIdx.x * 4;
    const int mtile = blockIdx.y;
    const int mrow  = mtile * 16 + (lane & 15);
    const int half  = lane >> 4;

    v8f acc[4] = {};
    const uint16_t* a0 = A + (size_t)mrow * lda + half * 8;
    const size_t bstride = (size_t)KS * 512;            // elements per N-tile
    const uint16_t* bp = Bsw + (size_t)nt0 * bstride + lane * 16;

    for (int ks = 0; ks < KS; ++ks) {
        Frag32 a;
        a.u4[0] = *(const uint4*)(a0 + ks * 32);
        a.u4[1] = *(const uint4*)(a0 + ks * 32 + 16);
        const uint16_t* bk = bp + (size_t)ks * 512;
        Frag32 b0, b1, b2, b3;
        b0.u4[0] = *(const uint4*)(bk);
        b0.u4[1] = *(const uint4*)(bk + 8);
        b1.u4[0] = *(const uint4*)(bk + bstride);
        b1.u4[1] = *(const uint4*)(bk + bstride + 8);
        b2.u4[0] = *(const uint4*)(bk + 2 * bstride);
        b2.u4[1] = *(const uint4*)(bk + 2 * bstride + 8);
        b3.u4[0] = *(const uint4*)(bk + 3 * bstride);
        b3.u4[1] = *(const uint4*)(bk + 3 * bstride + 8);
        acc[0] = __builtin_amdgcn_wmma_f32_16x16x32_bf16(
                     false, a.v, false, b0.v, (short)0, acc[0], false, false);
        acc[1] = __builtin_amdgcn_wmma_f32_16x16x32_bf16(
                     false, a.v, false, b1.v, (short)0, acc[1], false, false);
        acc[2] = __builtin_amdgcn_wmma_f32_16x16x32_bf16(
                     false, a.v, false, b2.v, (short)0, acc[2], false, false);
        acc[3] = __builtin_amdgcn_wmma_f32_16x16x32_bf16(
                     false, a.v, false, b3.v, (short)0, acc[3], false, false);
    }

    const int row0 = mtile * 16 + half * 8;
#pragma unroll
    for (int j = 0; j < 4; ++j) {
        const int col = (nt0 + j) * 16 + (lane & 15);
        if (col >= N) continue;                 // padded N tiles (z GEMM only)
        const float bv = bias[col];
#pragma unroll
        for (int r = 0; r < 8; ++r) {
            const int row = row0 + r;
            float v = acc[j][r] + bv;
            if (MODE == 1) {
                v = v > 0.0f ? v : 0.0f;
                Cbf[(size_t)row * N + col] = f2bf(v);
            } else if (MODE == 2) {
                Cf[(size_t)row * N + col] = sigm(v);
            } else if (MODE == 3) {
                Cf[(size_t)row * N + col] =
                    theme[(size_t)row * N + col] * v + hAdd[(size_t)row * N + col];
            } else {
                Cf[(size_t)row * N + col] = v;
            }
        }
    }
}

// ---------------- per-step elementwise: gates, cell, ring buffers, conv-A build ----------------
__global__ void sn_step(const float* __restrict__ z,
                        float* __restrict__ cbuf,
                        float* __restrict__ hRnn,
                        uint16_t* __restrict__ Az,
                        float* __restrict__ tmp_h,
                        float* __restrict__ tmp_dis,
                        uint16_t* __restrict__ Aconv,
                        uint16_t* __restrict__ meanA,
                        int t) {
    __shared__ float s_fm[LL], s_im[LL], s_ld[KK];
    const int b = blockIdx.x;
    const int slot = t % KK;

    if (threadIdx.x == 0) {
        const float* zb = z + (size_t)b * GG;
        float w0[LL], w1[LL], m0 = -1e30f, m1 = -1e30f;
        for (int j = 0; j < LL; ++j) {
            w0[j] = zb[j]; w1[j] = zb[LL + j];
            m0 = fmaxf(m0, w0[j]); m1 = fmaxf(m1, w1[j]);
        }
        float s0 = 0.f, s1 = 0.f;
        for (int j = 0; j < LL; ++j) {
            w0[j] = expf(w0[j] - m0); w1[j] = expf(w1[j] - m1);
            s0 += w0[j]; s1 += w1[j];
        }
        float run = 0.f, fmean = 0.f;
        for (int j = 0; j < LL; ++j) { run += w0[j] / s0; s_fm[j] = run; fmean += run; }
        float run1 = 0.f;
        for (int j = LL - 1; j >= 0; --j) { run1 += w1[j] / s1; s_im[j] = run1; }
        const float cur_dis = 1.0f - fmean / (float)LL;
        tmp_dis[slot * BB + b] = cur_dis;
        // local_dis = softmax over cumsum of window (oldest -> newest)
        float cs = 0.f, dv[KK], mx = -1e30f;
        for (int p = 0; p < KK; ++p) {
            int sp = (t + 1 + p) % KK;
            cs += tmp_dis[sp * BB + b];
            dv[p] = cs; mx = fmaxf(mx, cs);
        }
        float ssum = 0.f;
        for (int p = 0; p < KK; ++p) { dv[p] = expf(dv[p] - mx); ssum += dv[p]; }
        for (int p = 0; p < KK; ++p) s_ld[p] = dv[p] / ssum;
    }
    __syncthreads();

    const float* zb = z + (size_t)b * GG + 2 * LL;   // gate block, (4L x C) row-major
    for (int c = threadIdx.x; c < HH; c += blockDim.x) {
        const int lrow = c / CCH;
        const float fg = sigm(zb[c]);
        const float ig = sigm(zb[HH + c]);
        const float og = sigm(zb[2 * HH + c]);
        const float ci = tanhf(zb[3 * HH + c]);
        const float fmv = s_fm[lrow], imv = s_im[lrow], ov = fmv * imv;
        const float cl = cbuf[(size_t)b * HH + c];
        const float cn = ov * (fg * cl + ig * ci) + (fmv - ov) * cl + (imv - ov) * ci;
        const float hn = og * tanhf(cn);
        cbuf[(size_t)b * HH + c] = cn;
        hRnn[(size_t)b * HH + c] = hn;
        Az[(size_t)b * KA + 77 + c] = f2bf(hn);               // feed next z-GEMM
        tmp_h[((size_t)slot * BB + b) * HH + c] = hn;
        // build conv A-matrix (local_h * local_dis) and its mean over K
        float sum = 0.f;
        for (int p = 0; p < KK; ++p) {
            const int sp = (t + 1 + p) % KK;
            const float lh = tmp_h[((size_t)sp * BB + b) * HH + c] * s_ld[p];
            Aconv[(size_t)b * HK + (size_t)c * KK + p] = f2bf(lh);
            sum += lh;
        }
        meanA[(size_t)b * HH + c] = f2bf(sum * (1.0f / (float)KK));
    }
}

// ---------------- per-step output: sigmoid(rnn @ out_w + out_b) ----------------
__global__ void sn_outdot(const float* __restrict__ rnn,
                          const float* __restrict__ out_w,
                          const float* __restrict__ out_b,
                          float* __restrict__ out, int t) {
    __shared__ float red[256];
    const int b = blockIdx.x;
    float s = 0.f;
    for (int c = threadIdx.x; c < HH; c += 256)
        s += rnn[(size_t)b * HH + c] * out_w[c];
    red[threadIdx.x] = s;
    __syncthreads();
    for (int st = 128; st > 0; st >>= 1) {
        if ((int)threadIdx.x < st) red[threadIdx.x] += red[threadIdx.x + st];
        __syncthreads();
    }
    if (threadIdx.x == 0)
        out[(size_t)b * TT + t] = sigm(red[0] + out_b[0]);
}

// ---------------- host ----------------
extern "C" void kernel_launch(void* const* d_in, const int* in_sizes, int n_in,
                              void* d_out, int out_size, void* d_ws, size_t ws_size,
                              hipStream_t stream) {
    const float* x         = (const float*)d_in[0];
    const float* kernel_w  = (const float*)d_in[1];
    const float* kernel_b  = (const float*)d_in[2];
    const float* rec_w     = (const float*)d_in[3];
    const float* rec_b     = (const float*)d_in[4];
    const float* scale_w   = (const float*)d_in[5];
    const float* scale_b   = (const float*)d_in[6];
    const float* rescale_w = (const float*)d_in[7];
    const float* rescale_b = (const float*)d_in[8];
    const float* conv_w    = (const float*)d_in[9];
    const float* conv_b    = (const float*)d_in[10];
    const float* out_w     = (const float*)d_in[11];
    const float* out_b     = (const float*)d_in[12];
    float* out = (float*)d_out;

    // workspace carving (256B aligned)
    uint8_t* ws = (uint8_t*)d_ws;
    size_t off = 0;
    auto take = [&](size_t bytes) -> uint8_t* {
        uint8_t* p = ws + off;
        off = (off + bytes + 255) & ~(size_t)255;
        return p;
    };
    uint16_t* Wz_sw    = (uint16_t*)take((size_t)KA * NTZ * 16 * 2);  // 864 x 3136 bf16 (padded)
    uint16_t* conv_sw  = (uint16_t*)take((size_t)HK * HH * 2);        // 7680x768 bf16
    uint16_t* scale_sw = (uint16_t*)take((size_t)HH * SS * 2);
    uint16_t* resc_sw  = (uint16_t*)take((size_t)SS * HH * 2);
    float*    biasZ    = (float*)take((size_t)GG * 4);
    uint16_t* Az       = (uint16_t*)take((size_t)BB * KA * 2);
    float*    zbuf     = (float*)take((size_t)BB * GG * 4);
    float*    cbuf     = (float*)take((size_t)BB * HH * 4);
    float*    hRnn     = (float*)take((size_t)BB * HH * 4);
    float*    tmp_h    = (float*)take((size_t)KK * BB * HH * 4);
    float*    tmp_dis  = (float*)take((size_t)KK * BB * 4);
    uint16_t* Aconv    = (uint16_t*)take((size_t)BB * HK * 2);
    uint16_t* meanA    = (uint16_t*)take((size_t)BB * HH * 2);
    uint16_t* t1buf    = (uint16_t*)take((size_t)BB * SS * 2);
    float*    theme    = (float*)take((size_t)BB * HH * 4);
    float*    rnn      = (float*)take((size_t)BB * HH * 4);
    (void)ws_size; (void)n_in; (void)in_sizes; (void)out_size;

    // weight swizzle + bias fold (cheap vs. 400-step scan; deterministic)
    {
        int tot = NTZ * (KA / 32) * 512;         // Wz: 196*27*512 (padded N)
        sn_swizzleB<<<(tot + 255) / 256, 256, 0, stream>>>(kernel_w, rec_w, Wz_sw, KA / 32, tot, GG, 0);
        tot = (HH / 16) * (HK / 32) * 512;       // conv: 48*240*512
        sn_swizzleB<<<(tot + 255) / 256, 256, 0, stream>>>(conv_w, nullptr, conv_sw, HK / 32, tot, HH, 1);
        tot = (SS / 16) * (HH / 32) * 512;       // scale: 8*24*512
        sn_swizzleB<<<(tot + 255) / 256, 256, 0, stream>>>(scale_w, nullptr, scale_sw, HH / 32, tot, SS, 2);
        tot = (HH / 16) * (SS / 32) * 512;       // rescale: 48*4*512
        sn_swizzleB<<<(tot + 255) / 256, 256, 0, stream>>>(rescale_w, nullptr, resc_sw, SS / 32, tot, HH, 3);
        sn_biasz<<<(GG + 255) / 256, 256, 0, stream>>>(kernel_b, rec_b, biasZ);
    }

    // zero recurrent state (every call: deterministic under graph replay)
    sn_zero_f32<<<(BB * HH + 255) / 256, 256, 0, stream>>>(cbuf, BB * HH);
    sn_zero_f32<<<(KK * BB * HH + 255) / 256, 256, 0, stream>>>(tmp_h, KK * BB * HH);
    sn_zero_f32<<<(KK * BB + 255) / 256, 256, 0, stream>>>(tmp_dis, KK * BB);
    sn_zero_f32<<<(BB * KA / 2 + 255) / 256, 256, 0, stream>>>((float*)Az, BB * KA / 2);

    // sequential scan
    for (int t = 0; t < TT; ++t) {
        sn_loadx<<<BB, 128, 0, stream>>>(x, Az, t);
        sn_gemm_bf16<0><<<dim3(NTZ / 4, BB / 16), 32, 0, stream>>>(
            Az, KA, Wz_sw, biasZ, zbuf, nullptr, GG, KA / 32, nullptr, nullptr);
        sn_step<<<BB, 256, 0, stream>>>(zbuf, cbuf, hRnn, Az, tmp_h, tmp_dis, Aconv, meanA, t);
        sn_gemm_bf16<1><<<dim3(SS / 64, BB / 16), 32, 0, stream>>>(
            meanA, HH, scale_sw, scale_b, nullptr, t1buf, SS, HH / 32, nullptr, nullptr);
        sn_gemm_bf16<2><<<dim3(HH / 64, BB / 16), 32, 0, stream>>>(
            t1buf, SS, resc_sw, rescale_b, theme, nullptr, HH, SS / 32, nullptr, nullptr);
        sn_gemm_bf16<3><<<dim3(HH / 64, BB / 16), 32, 0, stream>>>(
            Aconv, HK, conv_sw, conv_b, rnn, nullptr, HH, HK / 32, theme, hRnn);
        sn_outdot<<<BB, 256, 0, stream>>>(rnn, out_w, out_b, out, t);
    }
}
